// MPNNLayer_60215441490190
// MI455X (gfx1250) — matrix operand order repaired and checked by reading, not compiled, not run on hardware
//
#include <hip/hip_runtime.h>
#include <hip/hip_bf16.h>

// Problem constants (match reference)
#define NNODE 20000
#define NEDGE 40000
#define NFEAT 64
#define EFEAT 32
#define OFEAT 64
#define HID   4096   // OF*NF : edge-MLP hidden width
#define OO    4096   // NF*OF : flattened per-edge weight matrix
#define BN_EPS 1e-5f

typedef __attribute__((ext_vector_type(16))) __bf16         v16bf;
typedef __attribute__((ext_vector_type(8)))  float          v8f;
typedef __attribute__((ext_vector_type(16))) unsigned short v16us;

__device__ __forceinline__ unsigned short f2bf(float f) {
    unsigned int u = __float_as_uint(f);
    unsigned int r = u + 0x7FFFu + ((u >> 16) & 1u);   // round-to-nearest-even
    return (unsigned short)(r >> 16);
}

// ---------------------------------------------------------------- utilities
__global__ void zero_kernel(float* __restrict__ p, int n) {
    int i = blockIdx.x * blockDim.x + threadIdx.x;
    if (i < n) p[i] = 0.0f;
}

// W2 [4096,4096] f32 row-major  ->  bf16, pre-swizzled into WMMA B-fragment
// lane-major order so a lane's 16 elements are 32 contiguous bytes:
//   n -> p = n>>7 (128-col pass), wv = (n>>4)&7 (wave), nl = n&15
//   k -> ku = k>>5 (k-step),      kk = k&31: lane = (kk>>4)*16 + nl, e = kk&15
//   dst = ((p*128 + ku)*4096) + wv*512 + lane*16 + e
__global__ void cvt_w2_swizzle_kernel(const float* __restrict__ W2,
                                      unsigned short* __restrict__ w2sw) {
    const int gid = blockIdx.x * 256 + threadIdx.x;   // = k*4096 + n
    const int k = gid >> 12, n = gid & 4095;
    const int p = n >> 7, wv = (n >> 4) & 7, nl = n & 15;
    const int ku = k >> 5, kk = k & 31;
    const int lane = ((kk >> 4) << 4) + nl, e = kk & 15;
    const size_t dst =
        ((size_t)(p * 128 + ku) << 12) + (wv << 9) + (lane << 4) + e;
    w2sw[dst] = f2bf(W2[gid]);
}

// --------------------------------------------- GEMM1: h = relu(efeat@W1+b1)
__global__ __launch_bounds__(256) void edge_mlp1_kernel(
    const float* __restrict__ efeat, const float* __restrict__ W1,
    const float* __restrict__ b1, unsigned short* __restrict__ hbf) {
    __shared__ float ef[EFEAT];
    const int e = blockIdx.x >> 4;
    const int c = ((blockIdx.x & 15) << 8) + threadIdx.x;
    if (threadIdx.x < EFEAT) ef[threadIdx.x] = efeat[e * EFEAT + threadIdx.x];
    __syncthreads();
    float acc = b1[c];
#pragma unroll
    for (int k = 0; k < EFEAT; ++k)
        acc = fmaf(ef[k], W1[k * HID + c], acc);
    hbf[(size_t)e * HID + c] = f2bf(fmaxf(acc, 0.0f));
}

// ----------------- Fused: We = h@W2+b2 ; msg = x[src]·We ; agg[dst] += msg
// Block: 256 threads (8 wave32), M-tile = 32 edges (two 16-row WMMA tiles).
// 32 passes of 128 columns (wave w owns cols [c0+16w, c0+16w+16), one i each).
// K in 8 chunks of 512 staged in LDS (A only); B fragments load straight from
// the pre-swizzled w2sw (L2-resident), 32 contiguous bytes per lane.
__global__ __launch_bounds__(256) void nnconv_fused_kernel(
    const unsigned short* __restrict__ hbf,   // [E, 4096] bf16 row-major
    const unsigned short* __restrict__ w2sw,  // swizzled [4096,4096] bf16
    const float* __restrict__ b2,             // [4096]
    const float* __restrict__ x,              // [N, 64]
    const int* __restrict__ src,
    const int* __restrict__ dst,
    float* __restrict__ agg) {                // [N, 64] (atomic accum)
    __shared__ __align__(16) unsigned short lds_a[32 * 512];   // 32 KB
    __shared__ float lds_msg[32 * 64];                         //  8 KB
    __shared__ float lds_x[32 * 64];                           //  8 KB
    __shared__ int   lds_src[32];

    const int tid  = threadIdx.x;
    const int wave = tid >> 5;
    const int lane = tid & 31;
    const int nl   = lane & 15;
    const int half = lane >> 4;
    const int e0   = blockIdx.x * 32;

    if (tid < 32) lds_src[tid] = src[e0 + tid];
    for (int i = tid; i < 32 * 64; i += 256) lds_msg[i] = 0.0f;
    __syncthreads();
    for (int i = tid; i < 32 * 64; i += 256)
        lds_x[i] = x[(size_t)lds_src[i >> 6] * NFEAT + (i & 63)];

    union AF { uint4 q[2]; v16us v; };

    for (int p = 0; p < 32; ++p) {
        const int c0 = p * 128;
        v8f acc0 = {};   // edges e0+0..15
        v8f acc1 = {};   // edges e0+16..31
        for (int kc = 0; kc < 8; ++kc) {            // 512-wide K chunks
            __syncthreads();
            // stage A chunk [32 x 512] : 8x 16B per thread, coalesced
#pragma unroll
            for (int i = tid * 8; i < 32 * 512; i += 256 * 8) {
                const int m = i >> 9, kk = i & 511;
                *reinterpret_cast<uint4*>(&lds_a[i]) =
                    *reinterpret_cast<const uint4*>(
                        &hbf[(size_t)(e0 + m) * HID + kc * 512 + kk]);
            }
            __syncthreads();
            const unsigned short* bbase =
                w2sw + ((size_t)(p * 128 + kc * 16) << 12) + (wave << 9) +
                (lane << 4);
#pragma unroll 2
            for (int t = 0; t < 16; ++t) {          // 32-wide k-steps
                // B fragment: 32 contiguous bytes per lane (2x b128)
                const v16us bu =
                    *reinterpret_cast<const v16us*>(bbase + ((size_t)t << 12));
                const v16bf b = __builtin_bit_cast(v16bf, bu);
                // A fragments: two contiguous 16B runs per lane (2x b128 each)
                const int koff = t * 32 + half * 16;
                AF a0, a1;
                a0.q[0] = *reinterpret_cast<const uint4*>(&lds_a[nl * 512 + koff]);
                a0.q[1] = *reinterpret_cast<const uint4*>(&lds_a[nl * 512 + koff + 8]);
                a1.q[0] = *reinterpret_cast<const uint4*>(&lds_a[(16 + nl) * 512 + koff]);
                a1.q[1] = *reinterpret_cast<const uint4*>(&lds_a[(16 + nl) * 512 + koff + 8]);
                acc0 = __builtin_amdgcn_wmma_f32_16x16x32_bf16(
                    false, __builtin_bit_cast(v16bf, a0.v), false, b,
                    (short)0, acc0, false, false);
                acc1 = __builtin_amdgcn_wmma_f32_16x16x32_bf16(
                    false, __builtin_bit_cast(v16bf, a1.v), false, b,
                    (short)0, acc1, false, false);
            }
        }
        // epilogue: column -> (i = col/64, o = col%64); contract with x[src]
        const int gcol = c0 + wave * 16 + nl;
        const int ii = gcol >> 6;
        const int oo = gcol & 63;
        const float bias = b2[gcol];
#pragma unroll
        for (int r = 0; r < 8; ++r) {
            const int m = r + half * 8;      // C/D row = r + 8*(lane>=16)
            atomicAdd(&lds_msg[m * 64 + oo],
                      lds_x[m * 64 + ii] * (acc0[r] + bias));
            atomicAdd(&lds_msg[(16 + m) * 64 + oo],
                      lds_x[(16 + m) * 64 + ii] * (acc1[r] + bias));
        }
    }
    __syncthreads();
    for (int i = tid; i < 32 * 64; i += 256) {
        const int m = i >> 6, o = i & 63;
        unsafeAtomicAdd(&agg[(size_t)dst[e0 + m] * OFEAT + o], lds_msg[i]);
    }
}

// ---------------- relu -> GRU(h0=0) -> residual -> pre + BN partial sums
__global__ __launch_bounds__(256) void node_update_kernel(
    const float* __restrict__ agg, const float* __restrict__ conv_b,
    const float* __restrict__ W_ih, const float* __restrict__ b_ih,
    const float* __restrict__ b_hh, const float* __restrict__ x,
    const float* __restrict__ W_res, const float* __restrict__ b_res,
    float* __restrict__ pre, float* __restrict__ bnsum,
    float* __restrict__ bnss) {
    __shared__ float nf[4 * 64];
    __shared__ float xr[4 * 64];
    __shared__ float nv[4 * 64];
    const int tid = threadIdx.x;
    const int ln = tid >> 6, o = tid & 63;
    const int node = blockIdx.x * 4 + ln;

    nf[tid] = fmaxf(agg[(size_t)node * OFEAT + o] + conv_b[o], 0.0f);
    xr[tid] = x[(size_t)node * NFEAT + o];
    __syncthreads();

    float gr = b_ih[o], gz = b_ih[64 + o], gn = b_ih[128 + o];
    float res = b_res[o];
#pragma unroll 4
    for (int k = 0; k < 64; ++k) {
        const float v = nf[ln * 64 + k];
        gr = fmaf(W_ih[(o)       * 64 + k], v, gr);
        gz = fmaf(W_ih[(64 + o)  * 64 + k], v, gz);
        gn = fmaf(W_ih[(128 + o) * 64 + k], v, gn);
        res = fmaf(xr[ln * 64 + k], W_res[k * OFEAT + o], res);
    }
    // h0 == 0 -> gh = b_hh ; z*h0 == 0
    const float r = 1.0f / (1.0f + expf(-(gr + b_hh[o])));
    const float z = 1.0f / (1.0f + expf(-(gz + b_hh[64 + o])));
    const float n = tanhf(gn + r * b_hh[128 + o]);
    const float nw = (1.0f - z) * n + res;

    pre[(size_t)node * OFEAT + o] = nw;
    nv[tid] = nw;
    __syncthreads();
    if (tid < 64) {
        const float a0 = nv[tid], a1 = nv[64 + tid],
                    a2 = nv[128 + tid], a3 = nv[192 + tid];
        unsafeAtomicAdd(&bnsum[tid], a0 + a1 + a2 + a3);
        unsafeAtomicAdd(&bnss[tid], a0 * a0 + a1 * a1 + a2 * a2 + a3 * a3);
    }
}

__global__ __launch_bounds__(256) void bn_finalize_kernel(
    const float* __restrict__ pre, const float* __restrict__ bnsum,
    const float* __restrict__ bnss, const float* __restrict__ gamma,
    const float* __restrict__ beta, float* __restrict__ out) {
    const int i = blockIdx.x * 256 + threadIdx.x;
    const int o = i & 63;
    const float mu = bnsum[o] * (1.0f / NNODE);
    const float var = bnss[o] * (1.0f / NNODE) - mu * mu;
    out[i] = (pre[i] - mu) * rsqrtf(var + BN_EPS) * gamma[o] + beta[o];
}

// --------------------------------------------------------------- dispatch
extern "C" void kernel_launch(void* const* d_in, const int* in_sizes, int n_in,
                              void* d_out, int out_size, void* d_ws,
                              size_t ws_size, hipStream_t stream) {
    const float* x      = (const float*)d_in[0];
    const float* efeat  = (const float*)d_in[1];
    const int*   src    = (const int*)d_in[2];
    const int*   dst    = (const int*)d_in[3];
    const float* W1     = (const float*)d_in[4];
    const float* b1     = (const float*)d_in[5];
    const float* W2     = (const float*)d_in[6];
    const float* b2     = (const float*)d_in[7];
    const float* conv_b = (const float*)d_in[8];
    const float* W_ih   = (const float*)d_in[9];
    const float* W_hh   = (const float*)d_in[10]; (void)W_hh; // h0==0
    const float* b_ih   = (const float*)d_in[11];
    const float* b_hh   = (const float*)d_in[12];
    const float* W_res  = (const float*)d_in[13];
    const float* b_res  = (const float*)d_in[14];
    const float* gamma  = (const float*)d_in[15];
    const float* beta   = (const float*)d_in[16];
    float* out = (float*)d_out;
    (void)in_sizes; (void)n_in; (void)out_size; (void)ws_size;

    // workspace carve-up (256B-aligned regions)
    char* ws = (char*)d_ws;
    size_t off = 0;
    auto carve = [&](size_t bytes) -> void* {
        void* p = ws + off;
        off = (off + bytes + 255) & ~(size_t)255;
        return p;
    };
    float* agg            = (float*)carve((size_t)NNODE * OFEAT * 4);
    float* pre            = (float*)carve((size_t)NNODE * OFEAT * 4);
    float* bnsum          = (float*)carve(128 * 4);          // [sum | sumsq]
    unsigned short* w2sw  = (unsigned short*)carve((size_t)HID * OO * 2);
    unsigned short* hbf   = (unsigned short*)carve((size_t)NEDGE * HID * 2);

    zero_kernel<<<(NNODE * OFEAT + 255) / 256, 256, 0, stream>>>(
        agg, NNODE * OFEAT);
    zero_kernel<<<1, 128, 0, stream>>>(bnsum, 128);
    cvt_w2_swizzle_kernel<<<(HID * OO) / 256, 256, 0, stream>>>(W2, w2sw);
    edge_mlp1_kernel<<<NEDGE * 16, 256, 0, stream>>>(efeat, W1, b1, hbf);
    nnconv_fused_kernel<<<NEDGE / 32, 256, 0, stream>>>(
        hbf, w2sw, b2, x, src, dst, agg);
    node_update_kernel<<<NNODE / 4, 256, 0, stream>>>(
        agg, conv_b, W_ih, b_ih, b_hh, x, W_res, b_res, pre, bnsum, bnsum + 64);
    bn_finalize_kernel<<<(NNODE * OFEAT) / 256, 256, 0, stream>>>(
        pre, bnsum, bnsum + 64, gamma, beta, out);
}
